// EnsembleQNetwork_32152125178527
// MI455X (gfx1250) — matrix operand order repaired and checked by reading, not compiled, not run on hardware
//
#include <hip/hip_runtime.h>

// Problem constants (match reference).
#define B_   16384
#define IN_  256
#define H_   512
#define OUT_ 32
#define E_   8
#define BM_  128   // batch rows per block (8 M-tiles of 16)

typedef __attribute__((ext_vector_type(16))) __bf16 v16bf;
typedef __attribute__((ext_vector_type(8)))  float  v8f;

union Frag {
    v16bf v;
    uint4 q[2];
};

// A fragment (16x32 bf16): lanes 0-15 take K-chunks {0..7,16..23}, lanes 16-31
// take {8..15,24..31} of row M=lane%16 -> two 16B loads at col kb+8*hi, +16.
__device__ __forceinline__ v16bf load_afrag(const __bf16* p) {
    Frag f;
    f.q[0] = *(const uint4*)(p);
    f.q[1] = *(const uint4*)(p + 16);
    return f.v;
}
// B fragment (32x16 bf16): lane holds col N = n0 + lane%16, rows K = kb + 8*hi + {0..7},
// kb + 16 + 8*hi + {0..7} (same K order as A) -> strided gather with pitch ld.
__device__ __forceinline__ v16bf load_bfrag(const __bf16* p, int ld) {
    v16bf v;
    #pragma unroll
    for (int i = 0; i < 8; ++i) {
        v[i]     = p[i * ld];
        v[8 + i] = p[(16 + i) * ld];
    }
    return v;
}

#define WMMA_BF16(Av, Bv, Cv) \
    __builtin_amdgcn_wmma_f32_16x16x32_bf16(false, (Av), false, (Bv), (short)0, (Cv), false, false)

// fp32 -> bf16 conversion prepass (grid-stride).
__global__ void cvt_f32_bf16(const float* __restrict__ src,
                             __bf16* __restrict__ dst, int n) {
    int i = blockIdx.x * blockDim.x + threadIdx.x;
    int stride = gridDim.x * blockDim.x;
    for (; i < n; i += stride) dst[i] = (__bf16)src[i];
}

// Fused 3-layer ensemble MLP. One block = 128 batch rows x 1 ensemble member.
// Weights are staged into LDS by 64-column slabs (read from L2 exactly once per
// block); layer 3 is fused into the layer-2 slab loop via a 16 KB H2-slab buffer.
__global__ __launch_bounds__(256)
void ensemble_mlp(const __bf16* __restrict__ Xb,
                  const __bf16* __restrict__ W1b,
                  const float*  __restrict__ b1,
                  const __bf16* __restrict__ W2b,
                  const float*  __restrict__ b2,
                  const __bf16* __restrict__ W3b,
                  const float*  __restrict__ b3,
                  float* __restrict__ out) {
    // 256 KB total:
    //   [0      , 131072): sH1   128x512 bf16 (128 KB)
    //   [131072 , 196608): sW    slab 512x64 bf16 max (64 KB)
    //   [196608 , 262144): sX    128x256 bf16 (64 KB, layer-1 only)
    //                      sH2s  128x64  bf16 (16 KB, layer-2, overlaps sX)
    //   [212992 , 245760): sW3   512x32  bf16 (32 KB, overlaps sX tail)
    __shared__ __align__(16) char smem[262144];
    __bf16* sH1  = (__bf16*)(smem);
    __bf16* sW   = (__bf16*)(smem + 131072);
    __bf16* sX   = (__bf16*)(smem + 196608);
    __bf16* sH2s = (__bf16*)(smem + 196608);
    __bf16* sW3  = (__bf16*)(smem + 212992);

    const int e    = blockIdx.y;
    const int mblk = blockIdx.x;
    const int tid  = threadIdx.x;
    const int wave = tid >> 5;
    const int lane = tid & 31;
    const int lm   = lane & 15;
    const int hi   = lane >> 4;

    const int mg   = wave & 3;   // M-group: 2 tiles at rows mg*32, mg*32+16
    const int ng   = wave >> 2;  // N-group: 32-col half of each 64-col slab
    const int mrow = mg * 32;

    const __bf16* W1e = W1b + (size_t)e * IN_ * H_;
    const __bf16* W2e = W2b + (size_t)e * H_  * H_;
    const __bf16* W3e = W3b + (size_t)e * H_  * OUT_;

    // ---- Stage X tile (128 x 256 bf16, contiguous 64 KB) ----
    {
        const uint4* s = (const uint4*)(Xb + (size_t)mblk * BM_ * IN_);
        uint4* d = (uint4*)sX;
        #pragma unroll 4
        for (int i = tid; i < BM_ * IN_ / 8; i += 256) d[i] = s[i];
    }
    __syncthreads();

    // ---- Layer 1: sH1 = relu(sX * W1 + b1); 8 slabs of 64 cols ----
    for (int s = 0; s < H_ / 64; ++s) {
        {   // stage W1[:, s*64 .. s*64+64) (256 rows, pitch H_) -> sW (pitch 64)
            const uint4* src = (const uint4*)(W1e + (size_t)s * 64);
            uint4* dst = (uint4*)sW;
            #pragma unroll 4
            for (int i = tid; i < IN_ * 8; i += 256)
                dst[i] = src[(size_t)(i >> 3) * (H_ / 8) + (i & 7)];
        }
        __syncthreads();

        v8f acc[2][2] = {};
        #pragma unroll
        for (int kb = 0; kb < IN_; kb += 32) {
            v16bf a0 = load_afrag(&sX[(mrow      + lm) * IN_ + kb + 8 * hi]);
            v16bf a1 = load_afrag(&sX[(mrow + 16 + lm) * IN_ + kb + 8 * hi]);
            v16bf bb0 = load_bfrag(&sW[(kb + 8 * hi) * 64 + ng * 32 + lm], 64);
            v16bf bb1 = load_bfrag(&sW[(kb + 8 * hi) * 64 + ng * 32 + 16 + lm], 64);
            acc[0][0] = WMMA_BF16(a0, bb0, acc[0][0]);
            acc[1][0] = WMMA_BF16(a1, bb0, acc[1][0]);
            acc[0][1] = WMMA_BF16(a0, bb1, acc[0][1]);
            acc[1][1] = WMMA_BF16(a1, bb1, acc[1][1]);
        }
        #pragma unroll
        for (int mi = 0; mi < 2; ++mi)
            #pragma unroll
            for (int ni = 0; ni < 2; ++ni) {
                const int ncol = s * 64 + ng * 32 + ni * 16 + lm;
                const float bias = b1[e * H_ + ncol];
                #pragma unroll
                for (int i = 0; i < 8; ++i) {
                    float v = acc[mi][ni][i] + bias;
                    v = v > 0.f ? v : 0.f;
                    sH1[(mrow + mi * 16 + i + 8 * hi) * H_ + ncol] = (__bf16)v;
                }
            }
        __syncthreads();
    }

    // ---- Stage W3 fully (512x32 bf16, contiguous 32 KB; sX is dead now) ----
    {
        const uint4* s2 = (const uint4*)W3e;
        uint4* d = (uint4*)sW3;
        #pragma unroll 4
        for (int i = tid; i < H_ * OUT_ / 8; i += 256) d[i] = s2[i];
    }

    // Layer-3 accumulators persist across slabs: 2 M-tiles x 1 out-tile (n0 = ng*16).
    v8f accO[2] = {};

    // ---- Layer 2 (+fused layer 3): 8 slabs of 64 cols ----
    for (int s = 0; s < H_ / 64; ++s) {
        {   // stage W2[:, s*64 .. s*64+64) (512 rows, pitch H_) -> sW (pitch 64)
            const uint4* src = (const uint4*)(W2e + (size_t)s * 64);
            uint4* dst = (uint4*)sW;
            #pragma unroll 4
            for (int i = tid; i < H_ * 8; i += 256)
                dst[i] = src[(size_t)(i >> 3) * (H_ / 8) + (i & 7)];
        }
        __syncthreads();

        v8f acc[2][2] = {};
        #pragma unroll 4
        for (int kb = 0; kb < H_; kb += 32) {
            v16bf a0 = load_afrag(&sH1[(mrow      + lm) * H_ + kb + 8 * hi]);
            v16bf a1 = load_afrag(&sH1[(mrow + 16 + lm) * H_ + kb + 8 * hi]);
            v16bf bb0 = load_bfrag(&sW[(kb + 8 * hi) * 64 + ng * 32 + lm], 64);
            v16bf bb1 = load_bfrag(&sW[(kb + 8 * hi) * 64 + ng * 32 + 16 + lm], 64);
            acc[0][0] = WMMA_BF16(a0, bb0, acc[0][0]);
            acc[1][0] = WMMA_BF16(a1, bb0, acc[1][0]);
            acc[0][1] = WMMA_BF16(a0, bb1, acc[0][1]);
            acc[1][1] = WMMA_BF16(a1, bb1, acc[1][1]);
        }
        // Epilogue: bias + relu -> H2 slab (pitch 64) in LDS.
        #pragma unroll
        for (int mi = 0; mi < 2; ++mi)
            #pragma unroll
            for (int ni = 0; ni < 2; ++ni) {
                const int ncg = s * 64 + ng * 32 + ni * 16 + lm;  // global col
                const int ncl = ng * 32 + ni * 16 + lm;           // col in slab
                const float bias = b2[e * H_ + ncg];
                #pragma unroll
                for (int i = 0; i < 8; ++i) {
                    float v = acc[mi][ni][i] + bias;
                    v = v > 0.f ? v : 0.f;
                    sH2s[(mrow + mi * 16 + i + 8 * hi) * 64 + ncl] = (__bf16)v;
                }
            }
        __syncthreads();  // H2 slab visible to all waves; sW reads complete

        // Fused layer 3: accO += H2slab(128x64) * W3[s*64 .. s*64+64)
        #pragma unroll
        for (int kk = 0; kk < 64; kk += 32) {
            v16bf a0 = load_afrag(&sH2s[(mrow      + lm) * 64 + kk + 8 * hi]);
            v16bf a1 = load_afrag(&sH2s[(mrow + 16 + lm) * 64 + kk + 8 * hi]);
            v16bf bb = load_bfrag(&sW3[(s * 64 + kk + 8 * hi) * OUT_ + ng * 16 + lm], OUT_);
            accO[0] = WMMA_BF16(a0, bb, accO[0]);
            accO[1] = WMMA_BF16(a1, bb, accO[1]);
        }
        __syncthreads();  // before next slab overwrites sW / sH2s
    }

    // ---- Output: bias3 + store fp32 [B, E, OUT] ----
    {
        const float bias = b3[e * OUT_ + ng * 16 + lm];
        #pragma unroll
        for (int mi = 0; mi < 2; ++mi)
            #pragma unroll
            for (int i = 0; i < 8; ++i) {
                const int row = mblk * BM_ + mrow + mi * 16 + i + 8 * hi;
                out[((size_t)row * E_ + e) * OUT_ + ng * 16 + lm] = accO[mi][i] + bias;
            }
    }
}

extern "C" void kernel_launch(void* const* d_in, const int* in_sizes, int n_in,
                              void* d_out, int out_size, void* d_ws, size_t ws_size,
                              hipStream_t stream) {
    const float* X  = (const float*)d_in[0];
    const float* W1 = (const float*)d_in[1];
    const float* b1 = (const float*)d_in[2];
    const float* W2 = (const float*)d_in[3];
    const float* b2 = (const float*)d_in[4];
    const float* W3 = (const float*)d_in[5];
    const float* b3 = (const float*)d_in[6];
    float* out = (float*)d_out;

    // Workspace layout: bf16 copies of X, W1, W2, W3 (~15 MB total).
    char* ws = (char*)d_ws;
    __bf16* Xb  = (__bf16*)ws;                 size_t o = (size_t)B_ * IN_ * 2;
    __bf16* W1b = (__bf16*)(ws + o);           o += (size_t)E_ * IN_ * H_ * 2;
    __bf16* W2b = (__bf16*)(ws + o);           o += (size_t)E_ * H_ * H_ * 2;
    __bf16* W3b = (__bf16*)(ws + o);

    cvt_f32_bf16<<<2048, 256, 0, stream>>>(X,  Xb,  B_ * IN_);
    cvt_f32_bf16<<<2048, 256, 0, stream>>>(W1, W1b, E_ * IN_ * H_);
    cvt_f32_bf16<<<2048, 256, 0, stream>>>(W2, W2b, E_ * H_ * H_);
    cvt_f32_bf16<<<512,  256, 0, stream>>>(W3, W3b, E_ * H_ * OUT_);

    dim3 grid(B_ / BM_, E_);
    ensemble_mlp<<<grid, 256, 0, stream>>>(Xb, W1b, b1, W2b, b2, W3b, b3, out);
}
